// EdgeNetwork_35399120454034
// MI455X (gfx1250) — compile-verified, matrix operand order checked by reading, not run on hardware
//
#include <hip/hip_runtime.h>

#define NODE_DIM 64
#define EDGE_DIM 16
#define IN_DIM   144
#define KPAD     160      // 144 padded to 5*32 for 16x16x32 bf16 WMMA
#define NEG_SLOPE 0.1f
#define LN_EPS    1e-5f

#define WAVES 4
#define TILES 4
#define EDGES_PER_BLOCK (WAVES*TILES*16)   // 256

typedef __attribute__((ext_vector_type(16))) __bf16 v16bf;
typedef __attribute__((ext_vector_type(8)))  __bf16 v8bf;
typedef __attribute__((ext_vector_type(2)))  __bf16 v2bf;
typedef __attribute__((ext_vector_type(8)))  float  v8f;

// packed f32x2 -> bf16x2 (native v_cvt_pk_bf16_f32 path via __bf16 casts)
__device__ __forceinline__ unsigned int pack2bf(float x, float y) {
  union { v2bf v; unsigned int u; } p;
  p.v[0] = (__bf16)x; p.v[1] = (__bf16)y;
  return p.u;
}

// Load one lane's 16 K-values of a 16x32 bf16 fragment
// (K = kc + half*8 + [0..7]  and  K = kc + 16 + half*8 + [0..7]) -> 2x ds_load_b128
__device__ __forceinline__ v16bf frag16(const __bf16* p) {
  v8bf lo = *reinterpret_cast<const v8bf*>(p);
  v8bf hi = *reinterpret_cast<const v8bf*>(p + 16);
  return __builtin_shufflevector(lo, hi, 0,1,2,3,4,5,6,7,8,9,10,11,12,13,14,15);
}

// wave-local LDS ordering point (DS ops are in-order per wave)
__device__ __forceinline__ void wave_lds_fence() {
  asm volatile("s_wait_dscnt 0" ::: "memory");
}

__device__ __forceinline__ float leaky(float v) {
  return fmaxf(v, 0.f) + NEG_SLOPE * fminf(v, 0.f);
}

__global__ __launch_bounds__(128) void edge_mlp_kernel(
    const float* __restrict__ nodef,
    const long long* __restrict__ eidx,
    const float* __restrict__ eattr,
    const float* __restrict__ W1, const float* __restrict__ b1,
    const float* __restrict__ g1, const float* __restrict__ be1,
    const float* __restrict__ W2, const float* __restrict__ b2,
    const float* __restrict__ g2, const float* __restrict__ be2,
    const float* __restrict__ W3, const float* __restrict__ b3,
    float* __restrict__ out, int E)
{
  __shared__ __align__(16) __bf16 sW1T[64*KPAD];   // W1^T [n][k], K zero-padded to 160
  __shared__ __align__(16) __bf16 sW2T[64*64];     // W2^T [n][k]
  __shared__ __align__(16) float sb1[64], sg1[64], sbe1[64], sb2[64], sg2[64], sbe2[64], sw3[64];
  __shared__ __align__(16) __bf16 sXT[WAVES][16*KPAD]; // per-wave x tile; f32 h tile aliases it
  __shared__ __align__(16) __bf16 sHB[WAVES][16*64];   // per-wave LN1 output (layer-2 A tile)

  const int tid = threadIdx.x;

  // ---- cooperative weight preload (transpose + bf16 convert) ----
  for (int i = tid; i < IN_DIM*64; i += 128) { int k=i>>6, n=i&63; sW1T[n*KPAD+k] = (__bf16)W1[i]; }
  for (int i = tid; i < (KPAD-IN_DIM)*64; i += 128) { int k=IN_DIM+(i>>6), n=i&63; sW1T[n*KPAD+k] = (__bf16)0.f; }
  for (int i = tid; i < 64*64; i += 128) { int k=i>>6, n=i&63; sW2T[n*64+k] = (__bf16)W2[i]; }
  if (tid < 64) {
    sb1[tid]=b1[tid]; sg1[tid]=g1[tid]; sbe1[tid]=be1[tid];
    sb2[tid]=b2[tid]; sg2[tid]=g2[tid]; sbe2[tid]=be2[tid]; sw3[tid]=W3[tid];
  }
  __syncthreads();

  const int w    = tid >> 5;
  const int lane = tid & 31;
  const int m16  = lane & 15;    // A row (edge in tile) / B,C,D column
  const int half = lane >> 4;    // K half-split for A/B fragments
  const int c0   = half*32;      // this lane's 32-column slice for LN passes

  __bf16* xt  = sXT[w];
  __bf16* htb = sHB[w];
  float*  htf = reinterpret_cast<float*>(sXT[w]);  // 16x64 f32 fits in dead x-tile

  const float bias3 = b3[0];

  // loop-invariant bias splats (cheap scalars only; no big register pins so the
  // scheduler keeps enough VGPR slack to software-pipeline the layer-1 loop)
  float bia1[4], bia2[4];
  #pragma unroll
  for (int nt = 0; nt < 4; ++nt) { bia1[nt] = sb1[m16+16*nt]; bia2[nt] = sb2[m16+16*nt]; }

  for (int t = 0; t < TILES; ++t) {
    const int e0 = blockIdx.x*EDGES_PER_BLOCK + w*(TILES*16) + t*16;
    const int e  = e0 + m16;
    const int ec = e < E ? e : (E-1);

    // ---- gather x tile: 16 edges x [src(64)|dst(64)|attr(16)|pad(16)] bf16 ----
    {
      long long si = eidx[ec];
      long long di = eidx[(long long)E + ec];
      const float4* s4 = reinterpret_cast<const float4*>(nodef + si*NODE_DIM + c0);
      const float4* d4 = reinterpret_cast<const float4*>(nodef + di*NODE_DIM + c0);
      unsigned int* xw = reinterpret_cast<unsigned int*>(xt + m16*KPAD);
      #pragma unroll
      for (int q = 0; q < 8; ++q) {
        float4 v = s4[q]; int c2 = (c0 + q*4) >> 1;
        xw[c2+0] = pack2bf(v.x, v.y); xw[c2+1] = pack2bf(v.z, v.w);
      }
      #pragma unroll
      for (int q = 0; q < 8; ++q) {
        float4 v = d4[q]; int c2 = (64 + c0 + q*4) >> 1;
        xw[c2+0] = pack2bf(v.x, v.y); xw[c2+1] = pack2bf(v.z, v.w);
      }
      if (half == 0) {
        const float4* a4 = reinterpret_cast<const float4*>(eattr + (long long)ec*EDGE_DIM);
        #pragma unroll
        for (int q = 0; q < 4; ++q) {
          float4 v = a4[q]; int c2 = (128 + q*4) >> 1;
          xw[c2+0] = pack2bf(v.x, v.y); xw[c2+1] = pack2bf(v.z, v.w);
        }
      } else {
        #pragma unroll
        for (int j = 0; j < 8; ++j) xw[(IN_DIM>>1) + j] = 0u;
      }
    }
    wave_lds_fence();

    // ---- layer 1: (16x160) @ (160x64), 20x v_wmma_f32_16x16x32_bf16 ----
    v8f acc[4];
    #pragma unroll
    for (int nt = 0; nt < 4; ++nt)
      #pragma unroll
      for (int r = 0; r < 8; ++r) acc[nt][r] = bia1[nt];
    #pragma unroll
    for (int kc = 0; kc < KPAD; kc += 32) {
      v16bf a = frag16(xt + m16*KPAD + kc + half*8);
      #pragma unroll
      for (int nt = 0; nt < 4; ++nt) {
        v16bf bb = frag16(sW1T + (m16 + 16*nt)*KPAD + kc + half*8);
        acc[nt] = __builtin_amdgcn_wmma_f32_16x16x32_bf16(false, a, false, bb,
                                                          (short)0, acc[nt], false, false);
      }
    }
    // D layout: lane%16 = N, VGPR r = row r + 8*half. Scatter to f32 LDS tile (transpose hop).
    #pragma unroll
    for (int nt = 0; nt < 4; ++nt)
      #pragma unroll
      for (int r = 0; r < 8; ++r)
        htf[(r + 8*half)*64 + m16 + 16*nt] = acc[nt][r];
    wave_lds_fence();

    // ---- LN1 + leaky -> bf16 A tile (single pass through registers) ----
    {
      const float4* hp = reinterpret_cast<const float4*>(htf + m16*64 + c0);
      float4 hv[8];
      float s = 0.f, ss = 0.f;
      #pragma unroll
      for (int q = 0; q < 8; ++q) {
        hv[q] = hp[q];
        s  += hv[q].x + hv[q].y + hv[q].z + hv[q].w;
        ss += hv[q].x*hv[q].x + hv[q].y*hv[q].y + hv[q].z*hv[q].z + hv[q].w*hv[q].w;
      }
      s  += __shfl_xor(s, 16, 32);
      ss += __shfl_xor(ss, 16, 32);
      float mu  = s * (1.f/64.f);
      float var = ss * (1.f/64.f) - mu*mu;
      float rs  = rsqrtf(var + LN_EPS);
      const float4* g4 = reinterpret_cast<const float4*>(sg1 + c0);
      const float4* o4 = reinterpret_cast<const float4*>(sbe1 + c0);
      unsigned int* hw = reinterpret_cast<unsigned int*>(htb + m16*64 + c0);
      #pragma unroll
      for (int q = 0; q < 8; ++q) {
        float4 g = g4[q], o = o4[q];
        float vx = leaky((hv[q].x - mu)*rs*g.x + o.x);
        float vy = leaky((hv[q].y - mu)*rs*g.y + o.y);
        float vz = leaky((hv[q].z - mu)*rs*g.z + o.z);
        float vw = leaky((hv[q].w - mu)*rs*g.w + o.w);
        hw[q*2+0] = pack2bf(vx, vy);
        hw[q*2+1] = pack2bf(vz, vw);
      }
    }
    wave_lds_fence();

    // ---- layer 2: (16x64) @ (64x64), 8x v_wmma ----
    v8f acc2[4];
    #pragma unroll
    for (int nt = 0; nt < 4; ++nt)
      #pragma unroll
      for (int r = 0; r < 8; ++r) acc2[nt][r] = bia2[nt];
    #pragma unroll
    for (int kc2 = 0; kc2 < 2; ++kc2) {
      v16bf a = frag16(htb + m16*64 + kc2*32 + half*8);
      #pragma unroll
      for (int nt = 0; nt < 4; ++nt) {
        v16bf bb = frag16(sW2T + (m16 + 16*nt)*64 + kc2*32 + half*8);
        acc2[nt] = __builtin_amdgcn_wmma_f32_16x16x32_bf16(false, a, false, bb,
                                                           (short)0, acc2[nt], false, false);
      }
    }
    #pragma unroll
    for (int nt = 0; nt < 4; ++nt)
      #pragma unroll
      for (int r = 0; r < 8; ++r)
        htf[(r + 8*half)*64 + m16 + 16*nt] = acc2[nt][r];
    wave_lds_fence();

    // ---- LN2 + leaky fused with final 64->1 projection ----
    {
      const float4* hp = reinterpret_cast<const float4*>(htf + m16*64 + c0);
      float4 hv[8];
      float s = 0.f, ss = 0.f;
      #pragma unroll
      for (int q = 0; q < 8; ++q) {
        hv[q] = hp[q];
        s  += hv[q].x + hv[q].y + hv[q].z + hv[q].w;
        ss += hv[q].x*hv[q].x + hv[q].y*hv[q].y + hv[q].z*hv[q].z + hv[q].w*hv[q].w;
      }
      s  += __shfl_xor(s, 16, 32);
      ss += __shfl_xor(ss, 16, 32);
      float mu  = s * (1.f/64.f);
      float var = ss * (1.f/64.f) - mu*mu;
      float rs  = rsqrtf(var + LN_EPS);
      const float4* g4 = reinterpret_cast<const float4*>(sg2 + c0);
      const float4* o4 = reinterpret_cast<const float4*>(sbe2 + c0);
      const float4* w4 = reinterpret_cast<const float4*>(sw3 + c0);
      float o = 0.f;
      #pragma unroll
      for (int q = 0; q < 8; ++q) {
        float4 g = g4[q], b = o4[q], wv = w4[q];
        o += leaky((hv[q].x - mu)*rs*g.x + b.x) * wv.x;
        o += leaky((hv[q].y - mu)*rs*g.y + b.y) * wv.y;
        o += leaky((hv[q].z - mu)*rs*g.z + b.z) * wv.z;
        o += leaky((hv[q].w - mu)*rs*g.w + b.w) * wv.w;
      }
      o += __shfl_xor(o, 16, 32);
      if (half == 0 && e < E) out[e] = o + bias3;
    }
    wave_lds_fence();  // next tile reuses xt/htf region
  }
}

extern "C" void kernel_launch(void* const* d_in, const int* in_sizes, int n_in,
                              void* d_out, int out_size, void* d_ws, size_t ws_size,
                              hipStream_t stream) {
  const float*     nodef = (const float*)d_in[0];
  const long long* eidx  = (const long long*)d_in[1];   // int64 per reference
  const float*     eattr = (const float*)d_in[2];
  const float* W1  = (const float*)d_in[3];
  const float* b1  = (const float*)d_in[4];
  const float* g1  = (const float*)d_in[5];
  const float* be1 = (const float*)d_in[6];
  const float* W2  = (const float*)d_in[7];
  const float* b2  = (const float*)d_in[8];
  const float* g2  = (const float*)d_in[9];
  const float* be2 = (const float*)d_in[10];
  const float* W3  = (const float*)d_in[11];
  const float* b3  = (const float*)d_in[12];
  float* out = (float*)d_out;

  const int E = in_sizes[1] / 2;   // edge_index is (2, E)
  const int blocks = (E + EDGES_PER_BLOCK - 1) / EDGES_PER_BLOCK;
  if (blocks > 0)
    edge_mlp_kernel<<<blocks, 128, 0, stream>>>(nodef, eidx, eattr,
                                                W1, b1, g1, be1,
                                                W2, b2, g2, be2,
                                                W3, b3, out, E);
}
